// NCODLoss_84688165142889
// MI455X (gfx1250) — compile-verified, matrix operand order checked by reading, not run on hardware
//
#include <hip/hip_runtime.h>

// Problem sizes (fixed by reference)
#define NB    16384          // batch
#define NC    1000           // classes
#define NCP   1024           // classes padded to 64 wmma tiles
#define ND    768            // embedding dim
#define EPS   1e-8f

typedef __attribute__((ext_vector_type(16))) __bf16 v16bf;
typedef __attribute__((ext_vector_type(8)))  __bf16 v8bf;
typedef __attribute__((ext_vector_type(8)))  float  v8f;

// ---------------------------------------------------------------------------
// Kernel 1: normalize centroids -> bf16, pad rows [1000,1024) with zeros.
// One wave per (padded) centroid row.
// ---------------------------------------------------------------------------
__global__ __launch_bounds__(256)
void ncod_centroid_norm(const float* __restrict__ centroids, __bf16* __restrict__ cnb) {
    const int wave = (blockIdx.x * blockDim.x + threadIdx.x) >> 5;
    const int lane = threadIdx.x & 31;
    if (wave >= NCP) return;
    __bf16* orow = cnb + (long)wave * ND;
    if (wave >= NC) {
        for (int j = lane; j < ND; j += 32) orow[j] = (__bf16)0.0f;
        return;
    }
    const float* row = centroids + (long)wave * ND;
    float ss = 0.0f;
    for (int j = lane; j < ND; j += 32) { float v = row[j]; ss += v * v; }
    #pragma unroll
    for (int off = 16; off > 0; off >>= 1) ss += __shfl_xor(ss, off, 32);
    const float scale = 1.0f / fmaxf(sqrtf(ss), EPS);
    for (int j = lane; j < ND; j += 32) orow[j] = (__bf16)(row[j] * scale);
}

// ---------------------------------------------------------------------------
// Kernel 2: cross-entropy. One wave per row of logits; per-block partial sums.
// ---------------------------------------------------------------------------
__global__ __launch_bounds__(256)
void ncod_ce(const float* __restrict__ logits, const int* __restrict__ targets,
             float* __restrict__ ce_part) {
    __shared__ float wsum[8];
    const int wib  = threadIdx.x >> 5;
    const int lane = threadIdx.x & 31;
    const int row  = blockIdx.x * 8 + wib;
    float ce = 0.0f;
    if (row < NB) {
        const float* lr = logits + (long)row * NC;
        float m = -3.4e38f;
        for (int j = lane; j < NC; j += 32) m = fmaxf(m, lr[j]);
        #pragma unroll
        for (int off = 16; off > 0; off >>= 1) m = fmaxf(m, __shfl_xor(m, off, 32));
        float s = 0.0f;
        for (int j = lane; j < NC; j += 32) s += __expf(lr[j] - m);
        #pragma unroll
        for (int off = 16; off > 0; off >>= 1) s += __shfl_xor(s, off, 32);
        if (lane == 0) {
            const float xt = lr[targets[row]];
            ce = -(xt - m - __logf(s));
        }
    }
    if (lane == 0) wsum[wib] = ce;
    __syncthreads();
    if (threadIdx.x == 0) {
        float acc = 0.0f;
        #pragma unroll
        for (int i = 0; i < 8; ++i) acc += wsum[i];
        ce_part[blockIdx.x] = acc;
    }
}

// ---------------------------------------------------------------------------
// Kernel 3: fused  normalize(emb) -> bf16 WMMA GEMM vs centroids -> softmax
// statistics. Block = 256 threads (8 waves) owns 16 embedding rows and all
// 1024 (padded) columns. sim tile lives entirely in LDS.
//   per-row outputs:  dist = sum_j (onehot_j - p_j)^2 = S2/S1^2 - 2*p_t + 1
//                     disc = 1 - p_t
// ---------------------------------------------------------------------------
__global__ __launch_bounds__(256, 1)
void ncod_gemm_softmax(const float* __restrict__ emb, const __bf16* __restrict__ cnb,
                       const int* __restrict__ targets, float* __restrict__ gpart) {
    __shared__ __align__(16) __bf16 A_lds[16][ND];      // 24 KB
    __shared__ __align__(16) float  sim[16][NCP];       // 64 KB
    __shared__ float rowc[16][2];

    const int tid  = threadIdx.x;
    const int wave = tid >> 5;          // 0..7
    const int lane = tid & 31;
    const int m0   = blockIdx.x * 16;   // first embedding row of this block

    // ---- Stage 1: normalize 16 rows -> bf16 in LDS (wave handles 2 rows) ----
    #pragma unroll
    for (int rr = 0; rr < 2; ++rr) {
        const int r = wave * 2 + rr;
        const float* er = emb + (long)(m0 + r) * ND;
        float ss = 0.0f;
        for (int j = lane; j < ND; j += 32) { float v = er[j]; ss += v * v; }
        #pragma unroll
        for (int off = 16; off > 0; off >>= 1) ss += __shfl_xor(ss, off, 32);
        const float scale = 1.0f / fmaxf(sqrtf(ss), EPS);
        for (int j = lane; j < ND; j += 32) A_lds[r][j] = (__bf16)(er[j] * scale);
    }
    __syncthreads();

    // ---- Stage 2: WMMA K-loop.  Wave covers cols [wave*128, wave*128+128) ----
    const int n16   = lane & 15;        // row(A) / col(B) within tile
    const int khalf = lane >> 4;        // K-half select (ISA 16-bit layout)

    v8f acc[8] = {};                    // 8 N-tiles of 16x16 f32
    const __bf16* bbase[8];
    #pragma unroll
    for (int t = 0; t < 8; ++t)
        bbase[t] = cnb + (long)(wave * 128 + t * 16 + n16) * ND + khalf * 8;

    for (int kc = 0; kc < ND; kc += 32) {
        const v8bf alo = *(const v8bf*)&A_lds[n16][kc + khalf * 8];
        const v8bf ahi = *(const v8bf*)&A_lds[n16][kc + 16 + khalf * 8];
        v16bf a;
        #pragma unroll
        for (int i = 0; i < 8; ++i) { a[i] = alo[i]; a[i + 8] = ahi[i]; }
        #pragma unroll
        for (int t = 0; t < 8; ++t) {
            const v8bf blo = *(const v8bf*)(bbase[t] + kc);
            const v8bf bhi = *(const v8bf*)(bbase[t] + kc + 16);
            v16bf b;
            #pragma unroll
            for (int i = 0; i < 8; ++i) { b[i] = blo[i]; b[i + 8] = bhi[i]; }
            acc[t] = __builtin_amdgcn_wmma_f32_16x16x32_bf16(
                false, a, false, b, (short)0, acc[t], false, false);
        }
    }

    // ---- Stage 3: spill C tiles to LDS (ISA C-layout: M = r + 8*khalf) ----
    const int crow = khalf * 8;
    #pragma unroll
    for (int t = 0; t < 8; ++t) {
        const int col = wave * 128 + t * 16 + n16;
        #pragma unroll
        for (int r = 0; r < 8; ++r) sim[crow + r][col] = acc[t][r];
    }
    __syncthreads();

    // ---- Stage 4: per-row softmax statistics (cols >= NC are padding) ----
    #pragma unroll
    for (int rr = 0; rr < 2; ++rr) {
        const int r = wave * 2 + rr;
        float m = -3.4e38f;
        for (int j = lane; j < NC; j += 32) m = fmaxf(m, sim[r][j]);
        #pragma unroll
        for (int off = 16; off > 0; off >>= 1) m = fmaxf(m, __shfl_xor(m, off, 32));
        float s1 = 0.0f, s2 = 0.0f;
        for (int j = lane; j < NC; j += 32) {
            const float e = __expf(sim[r][j] - m);
            s1 += e; s2 += e * e;
        }
        #pragma unroll
        for (int off = 16; off > 0; off >>= 1) {
            s1 += __shfl_xor(s1, off, 32);
            s2 += __shfl_xor(s2, off, 32);
        }
        if (lane == 0) {
            const int   tgt = targets[m0 + r];
            const float pt  = __expf(sim[r][tgt] - m) / s1;
            rowc[r][0] = s2 / (s1 * s1) - 2.0f * pt + 1.0f;  // sum_j (oh-p)^2
            rowc[r][1] = 1.0f - pt;                          // discount
        }
    }
    __syncthreads();
    if (tid == 0) {
        float d = 0.0f, g = 0.0f;
        #pragma unroll
        for (int r = 0; r < 16; ++r) { d += rowc[r][0]; g += rowc[r][1]; }
        gpart[blockIdx.x * 2 + 0] = d;
        gpart[blockIdx.x * 2 + 1] = g;
    }
}

// ---------------------------------------------------------------------------
// Kernel 4: deterministic final reduction + combine.
// ---------------------------------------------------------------------------
__global__ __launch_bounds__(256)
void ncod_finalize(const float* __restrict__ ce_part, const float* __restrict__ gpart,
                   const float* __restrict__ u, float* __restrict__ out) {
    __shared__ float red[256];
    const int t = threadIdx.x;
    float ce = 0.0f, d = 0.0f, g = 0.0f;
    for (int i = t; i < NB / 8; i += 256) ce += ce_part[i];          // 2048 partials
    for (int i = t; i < NB / 16; i += 256) {                         // 1024 partials
        d += gpart[2 * i + 0];
        g += gpart[2 * i + 1];
    }
    red[t] = ce; __syncthreads();
    for (int s = 128; s > 0; s >>= 1) { if (t < s) red[t] += red[t + s]; __syncthreads(); }
    const float ce_sum = red[0]; __syncthreads();
    red[t] = d; __syncthreads();
    for (int s = 128; s > 0; s >>= 1) { if (t < s) red[t] += red[t + s]; __syncthreads(); }
    const float d_sum = red[0]; __syncthreads();
    red[t] = g; __syncthreads();
    for (int s = 128; s > 0; s >>= 1) { if (t < s) red[t] += red[t + s]; __syncthreads(); }
    if (t == 0) {
        const float g_sum = red[0];
        out[0] = ce_sum / (float)NB
               + d_sum / ((float)NB * (float)NC)
               + (g_sum / (float)NB) * u[0];
    }
}

// ---------------------------------------------------------------------------
extern "C" void kernel_launch(void* const* d_in, const int* in_sizes, int n_in,
                              void* d_out, int out_size, void* d_ws, size_t ws_size,
                              hipStream_t stream) {
    const float* emb       = (const float*)d_in[0];   // [16384,768] f32
    const float* logits    = (const float*)d_in[1];   // [16384,1000] f32
    const int*   targets   = (const int*)  d_in[2];   // [16384] i32
    const float* centroids = (const float*)d_in[3];   // [1000,768] f32
    const float* u         = (const float*)d_in[4];   // [1] f32
    float*       out       = (float*)d_out;

    char* ws = (char*)d_ws;
    __bf16* cnb    = (__bf16*)ws;                              // 1024*768*2  = 1.5 MB
    float*  gpart  = (float*)(ws + (size_t)NCP * ND * 2);      // 1024*2 f32
    float*  cepart = (float*)(ws + (size_t)NCP * ND * 2 + NB / 16 * 2 * sizeof(float));

    ncod_centroid_norm<<<NCP / 8, 256, 0, stream>>>(centroids, cnb);
    ncod_ce<<<NB / 8, 256, 0, stream>>>(logits, targets, cepart);
    ncod_gemm_softmax<<<NB / 16, 256, 0, stream>>>(emb, cnb, targets, gpart);
    ncod_finalize<<<1, 256, 0, stream>>>(cepart, gpart, u, out);
}
